// MOELayerImpl_51462298141171
// MI455X (gfx1250) — matrix operand order repaired and checked by reading, not compiled, not run on hardware
//
#include <hip/hip_runtime.h>
#include <stdint.h>

typedef __attribute__((ext_vector_type(4)))  uint16_t u16x4;
typedef __attribute__((ext_vector_type(8)))  uint16_t u16x8;
typedef __attribute__((ext_vector_type(16))) uint16_t u16x16;
typedef __attribute__((ext_vector_type(16))) __bf16   v16bf;
typedef __attribute__((ext_vector_type(8)))  float    v8f;

#define S_TOK 8192
#define DMODEL 1024
#define NEXP 16
#define FF 4096
#define CAP 512

__device__ __forceinline__ uint16_t f2bf(float f) {
  return __builtin_bit_cast(uint16_t, (__bf16)f);   // native v_cvt f32->bf16 (RTNE)
}
__device__ __forceinline__ uint32_t pk2bf(float lo, float hi) {
  return (uint32_t)f2bf(lo) | ((uint32_t)f2bf(hi) << 16);
}

// ---------------- init: slot->token map only ----------------
__global__ void k_init(int* __restrict__ tfs) {
  int i = blockIdx.x * 256 + threadIdx.x;            // 0 .. 8191
  tfs[i] = -1;
}

// ---------------- gating (fp32, exact routing): logits + softmax + argmax ---------
// Block = 8 waves x 32 tokens; wave w computes K-chunk w (128 of 1024 terms) for all
// 32 tokens. Wg addresses are wave-uniform (readfirstlane) -> SMEM scalar loads.
__global__ __launch_bounds__(256) void k_gate(const float* __restrict__ x,
                                              const float* __restrict__ Wg,
                                              int* __restrict__ idx,
                                              float* __restrict__ gate) {
  __shared__ float red[8][32][16];                   // 16 KB
  int tid = threadIdx.x;
  int lane = tid & 31;
  int wu = __builtin_amdgcn_readfirstlane(tid >> 5); // wave id, provably uniform
  int tok0 = blockIdx.x * 32;
  int tok = tok0 + lane;
  const float* xr = x + (size_t)tok * DMODEL + wu * 128;
  const float* wg = Wg + wu * 128 * NEXP;            // uniform -> s_load
  float acc[NEXP];
#pragma unroll
  for (int e = 0; e < NEXP; ++e) acc[e] = 0.f;
  for (int m = 0; m < 128; ++m) {
    float xv = xr[m];
#pragma unroll
    for (int e = 0; e < NEXP; ++e) acc[e] += xv * wg[m * NEXP + e];
  }
#pragma unroll
  for (int e = 0; e < NEXP; ++e) red[wu][lane][e] = acc[e];
  __syncthreads();
  if (tid < 32) {
    float l[NEXP];
#pragma unroll
    for (int e = 0; e < NEXP; ++e) l[e] = 0.f;
    for (int w = 0; w < 8; ++w)
#pragma unroll
      for (int e = 0; e < NEXP; ++e) l[e] += red[w][tid][e];
    float mx = l[0]; int arg = 0;
#pragma unroll
    for (int e = 1; e < NEXP; ++e) { if (l[e] > mx) { mx = l[e]; arg = e; } }
    float sum = 0.f;
#pragma unroll
    for (int e = 0; e < NEXP; ++e) sum += __expf(l[e] - mx);
    idx[tok0 + tid]  = arg;
    gate[tok0 + tid] = 1.0f / sum;                   // softmax value at argmax
  }
}

// ---------------- ordered per-wave ranks + per-group histograms ----------------
__global__ void k_count(const int* __restrict__ idx, int* __restrict__ rank,
                        int* __restrict__ gcnt) {
  int tid = threadIdx.x;
  int s = blockIdx.x * 256 + tid;
  int lane = tid & 31;
  int g = s >> 5;                                    // 32-token group = one wave
  int my = idx[s];
  unsigned below = (1u << lane) - 1u;
  int r = 0;
#pragma unroll
  for (int e = 0; e < NEXP; ++e) {
    unsigned m = __builtin_amdgcn_ballot_w32(my == e);
    if (my == e)   r = __builtin_popcount(m & below);
    if (lane == e) gcnt[g * NEXP + e] = __builtin_popcount(m);
  }
  rank[s] = r;
}

// ---------------- serial per-expert scan over 256 groups ----------------
__global__ void k_scan(const int* __restrict__ gcnt, int* __restrict__ goff) {
  int e = threadIdx.x;
  if (e < NEXP) {
    int run = 0;
    for (int g = 0; g < 256; ++g) { goff[g * NEXP + e] = run; run += gcnt[g * NEXP + e]; }
  }
}

// ---------------- final locations, capacity drop, slot map ----------------
__global__ void k_loc(const int* __restrict__ idx, const int* __restrict__ rank,
                      const int* __restrict__ goff, const float* __restrict__ gate,
                      int* __restrict__ loc, float* __restrict__ scale,
                      int* __restrict__ tfs) {
  int s = blockIdx.x * 256 + threadIdx.x;
  int e = idx[s];
  int l = goff[(s >> 5) * NEXP + e] + rank[s];
  loc[s] = l;
  bool valid = l < CAP;
  scale[s] = valid ? gate[s] : 0.f;
  if (valid) tfs[e * CAP + l] = s;
}

// ------- scatter tokens to bf16 dispatch buffer; zero out-rows of dropped tokens ---
__global__ __launch_bounds__(256) void k_dispatch(const float* __restrict__ x,
                                                  const int* __restrict__ idx,
                                                  const int* __restrict__ loc,
                                                  uint32_t* __restrict__ disp32,
                                                  float* __restrict__ out) {
  int s = blockIdx.x;
  int l = loc[s];
  int t = threadIdx.x;                               // 256 threads x 4 elems
  if (l >= CAP) {                                    // dropped: emit zero output row
    float4 z; z.x = 0.f; z.y = 0.f; z.z = 0.f; z.w = 0.f;
    ((float4*)(out + (size_t)s * DMODEL))[t] = z;
    return;
  }
  int e = idx[s];
  const float4* src = (const float4*)(x + (size_t)s * DMODEL);
  uint32_t* dst = disp32 + ((size_t)e * CAP + l) * (DMODEL / 2);
  float4 v = src[t];
  dst[t * 2 + 0] = pk2bf(v.x, v.y);
  dst[t * 2 + 1] = pk2bf(v.z, v.w);
}

// ---------------- bf16 WMMA GEMM: C[512xNT] = A[512xKT] x bf16(B[KTxNT]) ----------------
// Block tile 128x128, 8 waves = 4 row-groups x 2 col-halves; wave tile 32x64.
// EPI 0: ReLU -> bf16 store to H.   EPI 1: scale-by-gate scatter to Out.
union Frag { u16x16 v; u16x8 h[2]; };

template<int KT, int NT, int EPI>
__global__ __launch_bounds__(256) void k_gemm(const uint16_t* __restrict__ A,
                                              const float* __restrict__ B,
                                              uint16_t* __restrict__ H,
                                              float* __restrict__ Out,
                                              const int* __restrict__ tfs,
                                              const float* __restrict__ scale) {
  __shared__ uint16_t Bs[128][48];                   // 32(K) x 128(N) tile, transposed, padded
  int e = blockIdx.z, br = blockIdx.y, bc = blockIdx.x;
  int tid = threadIdx.x, w = tid >> 5, lane = tid & 31;
  int wr = w & 3, wc = w >> 2;                       // 4x2 wave grid
  int laneM = lane & 15, laneH = lane >> 4;

  const uint16_t* Ae = A + (size_t)e * CAP * KT;
  const float*    Be = B + (size_t)e * KT * NT + (size_t)bc * 128;
  const u16x8* Arow0 = (const u16x8*)(Ae + (size_t)(br * 128 + wr * 32 + laneM) * KT);
  const u16x8* Arow1 = (const u16x8*)(Ae + (size_t)(br * 128 + wr * 32 + 16 + laneM) * KT);

  v8f acc[2][4];
#pragma unroll
  for (int a = 0; a < 2; ++a)
#pragma unroll
    for (int f = 0; f < 4; ++f)
#pragma unroll
      for (int i = 0; i < 8; ++i) acc[a][f][i] = 0.f;

  // staging coords: pair-pack two K rows per u32 store
  int kp = (tid >> 5) * 2;                           // even k base for j=0 (0..14)
  int cg = tid & 31;                                 // float4 column group
  // prefetch coords
  int pk = tid >> 3;                                 // 0..31
  int pc = (tid & 7) * 16;                           // 64B spacing

  const int KITER = KT / 32;
  for (int kb = 0; kb < KITER; ++kb) {
    int k0 = kb * 32;
    if (kb + 1 < KITER)
      __builtin_prefetch(Be + (size_t)(k0 + 32 + pk) * NT + pc, 0, 0);
    // stage fp32 B tile -> bf16 transposed LDS, two K rows packed per b32 store
#pragma unroll
    for (int j = 0; j < 2; ++j) {
      int k = kp + j * 16;
      float4 v0 = *(const float4*)(Be + (size_t)(k0 + k)     * NT + cg * 4);
      float4 v1 = *(const float4*)(Be + (size_t)(k0 + k + 1) * NT + cg * 4);
      uint32_t* bsp = (uint32_t*)&Bs[cg * 4][k];
      bsp[0 * 24] = pk2bf(v0.x, v1.x);               // row stride 48 u16 = 24 u32
      bsp[1 * 24] = pk2bf(v0.y, v1.y);
      bsp[2 * 24] = pk2bf(v0.z, v1.z);
      bsp[3 * 24] = pk2bf(v0.w, v1.w);
    }
    __syncthreads();
    // A fragments: lanes 0-15 hold K k0+0..7 / k0+16..23, lanes 16-31 hold +8
    Frag a0, a1;
    a0.h[0] = Arow0[(k0 >> 3) + laneH];
    a0.h[1] = Arow0[(k0 >> 3) + 2 + laneH];
    a1.h[0] = Arow1[(k0 >> 3) + laneH];
    a1.h[1] = Arow1[(k0 >> 3) + 2 + laneH];
    v16bf av0 = __builtin_bit_cast(v16bf, a0.v);
    v16bf av1 = __builtin_bit_cast(v16bf, a1.v);
#pragma unroll
    for (int f = 0; f < 4; ++f) {
      int col = wc * 64 + f * 16 + laneM;
      const u16x8* bp = (const u16x8*)&Bs[col][laneH * 8];
      Frag bfr;
      bfr.h[0] = bp[0]; bfr.h[1] = bp[2];
      v16bf bv = __builtin_bit_cast(v16bf, bfr.v);
      acc[0][f] = __builtin_amdgcn_wmma_f32_16x16x32_bf16(
          false, av0, false, bv, (short)0, acc[0][f], false, false);
      acc[1][f] = __builtin_amdgcn_wmma_f32_16x16x32_bf16(
          false, av1, false, bv, (short)0, acc[1][f], false, false);
    }
    __syncthreads();
  }

  if (EPI == 0) {
#pragma unroll
    for (int a = 0; a < 2; ++a) {
#pragma unroll
      for (int f = 0; f < 4; ++f) {
        int col = bc * 128 + wc * 64 + f * 16 + laneM;
        uint16_t* hp = H + ((size_t)e * CAP + br * 128 + wr * 32 + a * 16 + laneH * 8) * NT + col;
#pragma unroll
        for (int i = 0; i < 8; ++i) {
          float v = acc[a][f][i];
          hp[(size_t)i * NT] = f2bf(v > 0.f ? v : 0.f);
        }
      }
    }
  } else {
#pragma unroll
    for (int a = 0; a < 2; ++a) {
      int rbase = br * 128 + wr * 32 + a * 16 + laneH * 8;
      int tk[8]; float sc[8];
#pragma unroll
      for (int i = 0; i < 8; ++i) {
        int t = tfs[e * CAP + rbase + i];
        tk[i] = t;
        sc[i] = (t >= 0) ? scale[t] : 0.f;
      }
#pragma unroll
      for (int f = 0; f < 4; ++f) {
        int col = bc * 128 + wc * 64 + f * 16 + laneM;
#pragma unroll
        for (int i = 0; i < 8; ++i) {
          if (tk[i] >= 0) Out[(size_t)tk[i] * DMODEL + col] = acc[a][f][i] * sc[i];
        }
      }
    }
  }
}

extern "C" void kernel_launch(void* const* d_in, const int* in_sizes, int n_in,
                              void* d_out, int out_size, void* d_ws, size_t ws_size,
                              hipStream_t stream) {
  const float* x  = (const float*)d_in[0];
  const float* Wg = (const float*)d_in[1];
  const float* w1 = (const float*)d_in[2];
  const float* w2 = (const float*)d_in[3];
  float* out = (float*)d_out;
  char* ws = (char*)d_ws;

  int*      idx   = (int*)     (ws + 0);
  float*    gate  = (float*)   (ws + 32768);
  int*      rank  = (int*)     (ws + 65536);
  int*      loc   = (int*)     (ws + 98304);
  float*    scale = (float*)   (ws + 131072);
  int*      gcnt  = (int*)     (ws + 163840);
  int*      goff  = (int*)     (ws + 180224);
  int*      tfs   = (int*)     (ws + 196608);
  uint16_t* disp  = (uint16_t*)(ws + 229376);
  uint16_t* h     = (uint16_t*)(ws + 229376 + (size_t)NEXP * CAP * DMODEL * 2);

  k_init<<<NEXP * CAP / 256, 256, 0, stream>>>(tfs);
  k_gate<<<S_TOK / 32, 256, 0, stream>>>(x, Wg, idx, gate);
  k_count<<<S_TOK / 256, 256, 0, stream>>>(idx, rank, gcnt);
  k_scan<<<1, 32, 0, stream>>>(gcnt, goff);
  k_loc<<<S_TOK / 256, 256, 0, stream>>>(idx, rank, goff, gate, loc, scale, tfs);
  k_dispatch<<<S_TOK, 256, 0, stream>>>(x, idx, loc, (uint32_t*)disp, out);
  // h[e] = relu(disp[e] @ w1[e]) : 512 x 1024 x 4096
  k_gemm<DMODEL, FF, 0><<<dim3(FF / 128, CAP / 128, NEXP), 256, 0, stream>>>(
      disp, w1, h, nullptr, nullptr, nullptr);
  // out[token] = gate * (h[e] @ w2[e]) : 512 x 4096 x 1024, fused combine
  k_gemm<FF, DMODEL, 1><<<dim3(DMODEL / 128, CAP / 128, NEXP), 256, 0, stream>>>(
      h, w2, nullptr, out, tfs, scale);
}
